// SCNet_32306744000656
// MI455X (gfx1250) — matrix operand order repaired and checked by reading, not compiled, not run on hardware
//
#include <hip/hip_runtime.h>
#include <hip/hip_bf16.h>

typedef __attribute__((ext_vector_type(16))) _Float16 v16h;
typedef __attribute__((ext_vector_type(8)))  float    v8f;

// Issue one async global->LDS 16-byte copy (ASYNCcnt-tracked, GV mode).
__device__ __forceinline__ void async_copy_b128(unsigned lds_addr, const void* gptr) {
  asm volatile("global_load_async_to_lds_b128 %0, %1, off"
               :: "v"(lds_addr), "v"(gptr)
               : "memory");
}
__device__ __forceinline__ void wait_async0() {
  asm volatile("s_wait_asynccnt 0x0" ::: "memory");
}
__device__ __forceinline__ unsigned lds_addr_of(const void* p) {
  // Generic pointers to LDS carry the LDS byte offset in addr[31:0].
  return (unsigned)(unsigned long long)p;
}

// ---------------------------------------------------------------------------
// Kernel 1: Y[N,32] = [X | X*X] (N x 64) @ W (64 x 32), f16 WMMA, f32 accum.
// One wave per 16-row block, 8 waves (128 rows) per 256-thread block.
// W is staged into LDS PRE-SWIZZLED into B-fragment layout so each fragment
// is a single 32-byte contiguous LDS read (2x ds_load_b128) per lane:
//   frag f = ks*2 + nt ; lane L, elem e  ->  W[ks*32 + (e/8)*16 + (L/16)*8 + e%8][nt*16 + L%16]
// A-fragment (ISA 7.12.2, 16-bit A 16x32): lane L row m=L%16,
//   elem e -> K = (e/8)*16 + (L/16)*8 + (e%8)  (coalesced b128 global loads).
// ---------------------------------------------------------------------------
__global__ void __launch_bounds__(256)
poly_gemm_wmma(const float* __restrict__ X, const float* __restrict__ W,
               float* __restrict__ Y, int N) {
  __shared__ __align__(32) _Float16 sWf[4 * 32 * 16];   // 4 frags x 32 lanes x 16 halves
  const int tid = threadIdx.x;
  for (int i = tid; i < 2048; i += 256) {
    const int f  = i >> 9;          // 0..3 : (ks, nt) = (f>>1, f&1)
    const int L  = (i >> 4) & 31;   // lane
    const int e  = i & 15;          // element
    const int row = (f >> 1) * 32 + (e >> 3) * 16 + (L >> 4) * 8 + (e & 7);
    const int col = (f & 1) * 16 + (L & 15);
    sWf[i] = (_Float16)W[row * 32 + col];
  }
  __syncthreads();

  const int wave = tid >> 5, lane = tid & 31;
  const int half = lane >> 4, l16 = lane & 15;
  const int row0 = blockIdx.x * 128 + wave * 16;
  const int m = row0 + l16;

  const float* xr = X + (size_t)m * 32;
  float xa[16];
#pragma unroll
  for (int e = 0; e < 8; ++e) {
    xa[e]     = xr[half * 8 + e];        // K 0..15  (first power)
    xa[8 + e] = xr[16 + half * 8 + e];   // K 16..31
  }
  v16h a0, a1;   // k-step 0: X ; k-step 1: X*X (same columns, squared in regs)
#pragma unroll
  for (int e = 0; e < 16; ++e) {
    a0[e] = (_Float16)xa[e];
    a1[e] = (_Float16)(xa[e] * xa[e]);
  }

  const v16h* Wf = (const v16h*)sWf;
  v16h b00 = Wf[0 * 32 + lane];   // ks0, n-tile 0
  v16h b01 = Wf[1 * 32 + lane];   // ks0, n-tile 1
  v16h b10 = Wf[2 * 32 + lane];   // ks1, n-tile 0
  v16h b11 = Wf[3 * 32 + lane];   // ks1, n-tile 1

  v8f acc0 = {}, acc1 = {};
  acc0 = __builtin_amdgcn_wmma_f32_16x16x32_f16(false, a0, false, b00, (short)0, acc0, false, false);
  acc0 = __builtin_amdgcn_wmma_f32_16x16x32_f16(false, a1, false, b10, (short)0, acc0, false, false);
  acc1 = __builtin_amdgcn_wmma_f32_16x16x32_f16(false, a0, false, b01, (short)0, acc1, false, false);
  acc1 = __builtin_amdgcn_wmma_f32_16x16x32_f16(false, a1, false, b11, (short)0, acc1, false, false);

#pragma unroll
  for (int r = 0; r < 8; ++r) {
    Y[(size_t)(row0 + half * 8 + r) * 32 + l16]      = acc0[r];
    Y[(size_t)(row0 + half * 8 + r) * 32 + 16 + l16] = acc1[r];
  }
}

// ---------------------------------------------------------------------------
// Kernel 2: COO SPMM scatter:  Acc[rows[i], :] += vals[i] * Xin[cols[i], :]
// One wave per nnz, lane = feature column (F=32 == wave32). Atomic f32 adds
// fuse the multi-operator sums (n2n+e2n etc.) into shared accumulators.
// ---------------------------------------------------------------------------
__global__ void __launch_bounds__(256)
spmm_scatter(const int* __restrict__ rows, const int* __restrict__ cols,
             const float* __restrict__ vals, const float* __restrict__ Xin,
             float* __restrict__ Acc, int nnz) {
  const int i = blockIdx.x * 8 + (threadIdx.x >> 5);
  if (i >= nnz) return;
  const int lane = threadIdx.x & 31;
  const int r = rows[i];
  const int c = cols[i];
  const float v = vals[i];
  atomicAdd(&Acc[(size_t)r * 32 + lane], v * Xin[(size_t)c * 32 + lane]);
}

// ---------------------------------------------------------------------------
// Kernel 3: per-graph sort-pool (stable top-k by col 31 of scale*relu(A)) +
// conv1d(kernel=stride=64, CO=16) + relu. One 256-thread block per graph.
// Output layout matches reshape(b, CO, k): xcat[g, outOff + co*k + j].
// ---------------------------------------------------------------------------
__global__ void __launch_bounds__(256)
sortpool_conv(const float* __restrict__ X, const float* __restrict__ A, float scale,
              int npg, int k, const float* __restrict__ convW,
              float* __restrict__ xcat, int outOff) {
  __shared__ float sv[4096];
  __shared__ float rv[256];
  __shared__ int   ri[256];
  __shared__ int   sel[64];
  const int g = blockIdx.x, tid = threadIdx.x;
  const float* Xg = X + (size_t)g * npg * 32;
  const float* Ag = A + (size_t)g * npg * 32;

  for (int j = tid; j < npg; j += 256) {
    float a = Ag[(size_t)j * 32 + 31];
    sv[j] = scale * (a > 0.f ? a : 0.f);
  }
  __syncthreads();

  for (int it = 0; it < k; ++it) {
    float bv = -3.4e38f; int bi = 0x7fffffff;
    for (int j = tid; j < npg; j += 256) {
      float v = sv[j];
      if (v > bv) { bv = v; bi = j; }   // strict > keeps lowest index on ties
    }
    rv[tid] = bv; ri[tid] = bi;
    __syncthreads();
    for (int s = 128; s > 0; s >>= 1) {
      if (tid < s) {
        float ov = rv[tid + s]; int oi = ri[tid + s];
        if (ov > rv[tid] || (ov == rv[tid] && oi < ri[tid])) { rv[tid] = ov; ri[tid] = oi; }
      }
      __syncthreads();
    }
    if (tid == 0) { sel[it] = ri[0]; sv[ri[0]] = -3.4e38f; }
    __syncthreads();
  }

  // conv1d over concatenated row [X_row(32) | scale*relu(A_row)(32)]
  for (int o = tid; o < 16 * k; o += 256) {
    const int co = o / k, j = o % k;
    const int row = sel[j];
    const float* w = convW + co * 64;
    float s = 0.f;
#pragma unroll 8
    for (int c2 = 0; c2 < 32; ++c2) s += Xg[(size_t)row * 32 + c2] * w[c2];
#pragma unroll 8
    for (int c2 = 0; c2 < 32; ++c2) {
      float a = Ag[(size_t)row * 32 + c2];
      s += (scale * (a > 0.f ? a : 0.f)) * w[32 + c2];
    }
    xcat[(size_t)g * 1920 + outOff + o] = s > 0.f ? s : 0.f;
  }
}

// ---------------------------------------------------------------------------
// Kernel 4: MLP  out = relu(x @ W1[1920,128]) @ Wout[128,2],  x: [64,1920].
// Single block, 8 waves: wave w owns hidden-col tile n0=w*16. The 32x128 f32
// K-slab of W1 per step is CONTIGUOUS 16KB -> double-buffered into LDS with
// global_load_async_to_lds_b128 (issue slab k+1, WMMA on slab k,
// s_wait_asynccnt 0 + barrier). Converts 512B-strided dword traffic into
// fully coalesced async copies.
// ---------------------------------------------------------------------------
__global__ void __launch_bounds__(256)
mlp_wmma(const float* __restrict__ xcat, const float* __restrict__ W1,
         const float* __restrict__ Wout, float* __restrict__ out) {
  __shared__ __align__(16) float sb[2][32 * 128];   // double-buffered W1 K-slab
  __shared__ float h[64 * 128];
  const int tid = threadIdx.x, wave = tid >> 5, lane = tid & 31;
  const int half = lane >> 4, l16 = lane & 15;
  const int n0 = wave * 16;

  // Each thread stages 64B (4 x b128) of the 16KB slab.
  auto stage = [&](int kb, int buf) {
    const float* src = W1 + (size_t)kb * 4096 + tid * 16;
    const unsigned dst = lds_addr_of(&sb[buf][tid * 16]);
#pragma unroll
    for (int q = 0; q < 4; ++q)
      async_copy_b128(dst + q * 16, src + q * 4);
  };

  stage(0, 0);
  wait_async0();
  __syncthreads();

  v8f c0 = {}, c1 = {}, c2 = {}, c3 = {};
  for (int kb = 0; kb < 60; ++kb) {
    const int cur = kb & 1;
    if (kb + 1 < 60) stage(kb + 1, cur ^ 1);

    const float* slab = sb[cur];
    v16h b;
#pragma unroll
    for (int e = 0; e < 16; ++e)
      b[e] = (_Float16)slab[((e >> 3) * 16 + half * 8 + (e & 7)) * 128 + n0 + l16];

    const int kbase = kb * 32;
#pragma unroll
    for (int mt = 0; mt < 4; ++mt) {
      const int m = mt * 16 + l16;
      v16h a;
#pragma unroll
      for (int e = 0; e < 8; ++e) {
        a[e]     = (_Float16)xcat[(size_t)m * 1920 + kbase + half * 8 + e];
        a[8 + e] = (_Float16)xcat[(size_t)m * 1920 + kbase + 16 + half * 8 + e];
      }
      if      (mt == 0) c0 = __builtin_amdgcn_wmma_f32_16x16x32_f16(false, a, false, b, (short)0, c0, false, false);
      else if (mt == 1) c1 = __builtin_amdgcn_wmma_f32_16x16x32_f16(false, a, false, b, (short)0, c1, false, false);
      else if (mt == 2) c2 = __builtin_amdgcn_wmma_f32_16x16x32_f16(false, a, false, b, (short)0, c2, false, false);
      else              c3 = __builtin_amdgcn_wmma_f32_16x16x32_f16(false, a, false, b, (short)0, c3, false, false);
    }
    wait_async0();
    __syncthreads();
  }

#pragma unroll
  for (int r = 0; r < 8; ++r) {
    float v;
    v = c0[r]; h[( 0 + half * 8 + r) * 128 + n0 + l16] = v > 0.f ? v : 0.f;
    v = c1[r]; h[(16 + half * 8 + r) * 128 + n0 + l16] = v > 0.f ? v : 0.f;
    v = c2[r]; h[(32 + half * 8 + r) * 128 + n0 + l16] = v > 0.f ? v : 0.f;
    v = c3[r]; h[(48 + half * 8 + r) * 128 + n0 + l16] = v > 0.f ? v : 0.f;
  }
  __syncthreads();

  if (tid < 128) {
    const int gg = tid >> 1, o = tid & 1;
    float s = 0.f;
#pragma unroll 8
    for (int j = 0; j < 128; ++j) s += h[gg * 128 + j] * Wout[j * 2 + o];
    out[gg * 2 + o] = s;
  }
}

// ---------------------------------------------------------------------------
// Host launcher. Input index map (setup_inputs dict order):
//  0-2 L0, 3-5 L1, 6-8 L2, 9-11 D1invB1, 12-14 D2B1TD1inv, 15-17 B2TD2inv,
//  18-20 B2D3 (rows,cols,vals each), 21 X0, 22 X1, 23 X2,
//  24 W_n2n, 25 W_n2e, 26 W_e2e, 27 W_e2n, 28 W_e2t, 29 W_t2e, 30 W_t2t,
//  31-33 conv_w_{nodes,edges,tri}, 34 mlp_w1, 35 mlp_w_out, 36 batch_size.
// ---------------------------------------------------------------------------
extern "C" void kernel_launch(void* const* d_in, const int* in_sizes, int n_in,
                              void* d_out, int out_size, void* d_ws, size_t ws_size,
                              hipStream_t stream) {
  const float* X0 = (const float*)d_in[21];
  const float* X1 = (const float*)d_in[22];
  const float* X2 = (const float*)d_in[23];
  const int N0 = in_sizes[21] / 32;
  const int N1 = in_sizes[22] / 32;
  const int N2 = in_sizes[23] / 32;
  const int B  = 64;

  // Workspace (floats): A0[N0*32] A1[N1*32] A2[N2*32] tmp[N1*32] xcat[64*1920]
  float* A0   = (float*)d_ws;
  float* A1   = A0 + (size_t)N0 * 32;
  float* A2   = A1 + (size_t)N1 * 32;
  float* tmp  = A2 + (size_t)N2 * 32;
  float* xcat = tmp + (size_t)N1 * 32;

  hipMemsetAsync(A0, 0, (size_t)N0 * 32 * sizeof(float), stream);
  hipMemsetAsync(A1, 0, (size_t)N1 * 32 * sizeof(float), stream);
  hipMemsetAsync(A2, 0, (size_t)N2 * 32 * sizeof(float), stream);

  auto gemm = [&](const float* X, int widx, int N) {
    poly_gemm_wmma<<<N / 128, 256, 0, stream>>>(X, (const float*)d_in[widx], tmp, N);
  };
  auto spmm = [&](int base, float* Acc) {
    const int nnz = in_sizes[base + 2];
    spmm_scatter<<<(nnz + 7) / 8, 256, 0, stream>>>(
        (const int*)d_in[base], (const int*)d_in[base + 1],
        (const float*)d_in[base + 2], tmp, Acc, nnz);
  };

  // A0 <- n2n + e2n ; A1 <- n2e + e2e + t2e ; A2 <- e2t + t2t
  gemm(X0, 24, N0); spmm(0,  A0);   // n2n: L0         @ (X0p @ W_n2n)
  gemm(X1, 27, N1); spmm(9,  A0);   // e2n: D1invB1    @ (X1p @ W_e2n)
  gemm(X0, 25, N0); spmm(12, A1);   // n2e: D2B1TD1inv @ (X0p @ W_n2e)
  gemm(X1, 26, N1); spmm(3,  A1);   // e2e: L1         @ (X1p @ W_e2e)
  gemm(X2, 29, N2); spmm(18, A1);   // t2e: B2D3       @ (X2p @ W_t2e)
  gemm(X1, 28, N1); spmm(15, A2);   // e2t: B2TD2inv   @ (X1p @ W_e2t)
  gemm(X2, 30, N2); spmm(6,  A2);   // t2t: L2         @ (X2p @ W_t2t)

  sortpool_conv<<<B, 256, 0, stream>>>(X0, A0, 0.5f,        N0 / B, 30,
                                       (const float*)d_in[31], xcat, 0);
  sortpool_conv<<<B, 256, 0, stream>>>(X1, A1, 1.0f / 3.0f, N1 / B, 60,
                                       (const float*)d_in[32], xcat, 480);
  sortpool_conv<<<B, 256, 0, stream>>>(X2, A2, 0.5f,        N2 / B, 30,
                                       (const float*)d_in[33], xcat, 1440);

  mlp_wmma<<<1, 256, 0, stream>>>(xcat, (const float*)d_in[34],
                                  (const float*)d_in[35], (float*)d_out);
}
